// EdgeClassifierGNN_54820962566504
// MI455X (gfx1250) — matrix operand order repaired and checked by reading, not compiled, not run on hardware
//
#include <hip/hip_runtime.h>
#include <hip/hip_bf16.h>

#define NN 50000
#define NE 800000
#define DIM 256
#define KC 512   // concat K (agg|own) and (src|dst)

typedef __attribute__((ext_vector_type(16))) __bf16 v16bf;
typedef __attribute__((ext_vector_type(8)))  float  v8f;
typedef int v4i_vec __attribute__((vector_size(16)));

union FragBF { uint4 q[2]; v16bf v; };

#if defined(__HIP_DEVICE_COMPILE__) && \
    __has_builtin(__builtin_amdgcn_global_load_async_to_lds_b128) && \
    __has_builtin(__builtin_amdgcn_s_wait_asynccnt)
#define USE_ASYNC_LDS 1
#else
#define USE_ASYNC_LDS 0
#endif

// Stage one 64KB weight K-chunk: WT[n][c*128 .. c*128+127] -> sW[n*128 + kl]
__device__ __forceinline__ void stage_chunk(const __hip_bfloat16* __restrict__ WT,
                                            __hip_bfloat16* __restrict__ sW,
                                            int tid, int c) {
    const uint4* g = (const uint4*)WT;
    uint4* s = (uint4*)sW;
#if USE_ASYNC_LDS
    for (int i = tid; i < 256 * 16; i += 256) {
        int n = i >> 4, j = i & 15;
        __builtin_amdgcn_global_load_async_to_lds_b128(
            (__attribute__((address_space(1))) v4i_vec*)(g + n * 64 + c * 16 + j),
            (__attribute__((address_space(3))) v4i_vec*)(s + n * 16 + j),
            0, 0);
    }
    __builtin_amdgcn_s_wait_asynccnt(0);
#else
    for (int i = tid; i < 256 * 16; i += 256) {
        int n = i >> 4, j = i & 15;
        s[n * 16 + j] = g[n * 64 + c * 16 + j];
    }
#endif
}

// ---------------- degree ----------------
__global__ void deg_kernel(const int* __restrict__ dst, float* __restrict__ deg, int E) {
    int e = blockIdx.x * blockDim.x + threadIdx.x;
    if (e < E) atomicAdd(deg + dst[e], 1.0f);
}

// ---------------- scatter-add feat[src] -> agg[dst] ----------------
__global__ void scatter_kernel(const float* __restrict__ feat,
                               const int* __restrict__ src,
                               const int* __restrict__ dst,
                               float* __restrict__ agg, int E) {
    long long idx = (long long)blockIdx.x * blockDim.x + threadIdx.x;
    long long e = idx >> 6;
    int f = (int)(idx & 63) * 4;
    if (e < E) {
        long long s = src[e], d = dst[e];
        float4 v = *(const float4*)(feat + s * DIM + f);
        float* p = agg + d * DIM + f;
        atomicAdd(p + 0, v.x); atomicAdd(p + 1, v.y);
        atomicAdd(p + 2, v.z); atomicAdd(p + 3, v.w);
    }
}

// dstT[n*Ktot + k] = bf16( k<K1 ? A[k*N+n] : B[(k-K1)*N+n] )  (N-major transpose)
__global__ void pack_wt(const float* __restrict__ A, const float* __restrict__ B,
                        __hip_bfloat16* __restrict__ dstT, int K1, int Ktot, int N) {
    int idx = blockIdx.x * blockDim.x + threadIdx.x;
    if (idx >= N * Ktot) return;
    int n = idx / Ktot, k = idx % Ktot;
    float v = (k < K1) ? A[k * N + n] : B[(k - K1) * N + n];
    dstT[n * Ktot + k] = __float2bfloat16(v);
}

// abf[n, 0:256] = bf16(agg[n]/max(deg,1)); abf[n, 256:512] = bf16(own[n])
__global__ void build_a(const float* __restrict__ agg, const float* __restrict__ deg,
                        const float* __restrict__ own, __hip_bfloat16* __restrict__ abf, int N) {
    long long idx = (long long)blockIdx.x * blockDim.x + threadIdx.x;
    long long n = idx >> 6;
    int f = (int)(idx & 63) * 4;
    if (n >= N) return;
    float inv = 1.0f / fmaxf(deg[n], 1.0f);
    float4 av = *(const float4*)(agg + n * DIM + f);
    float4 ov = *(const float4*)(own + n * DIM + f);
    __hip_bfloat16* r = abf + n * KC;
    r[f + 0] = __float2bfloat16(av.x * inv);
    r[f + 1] = __float2bfloat16(av.y * inv);
    r[f + 2] = __float2bfloat16(av.z * inv);
    r[f + 3] = __float2bfloat16(av.w * inv);
    r[DIM + f + 0] = __float2bfloat16(ov.x);
    r[DIM + f + 1] = __float2bfloat16(ov.y);
    r[DIM + f + 2] = __float2bfloat16(ov.z);
    r[DIM + f + 3] = __float2bfloat16(ov.w);
}

// ------------- node GEMM: out = relu(A[M,512] @ W[512,256] + bias) -------------
// WT is N-major [256][512] bf16. LDS stages one 128-wide K chunk (64KB).
__global__ __launch_bounds__(256) void gemm_node(
    const __hip_bfloat16* __restrict__ A,
    const __hip_bfloat16* __restrict__ WT,
    const float* __restrict__ bias,
    float* __restrict__ outF, __hip_bfloat16* __restrict__ outB, int M)
{
    __shared__ __hip_bfloat16 sW[256 * 128];   // 64 KB chunk: sW[n*128 + klocal]
    const int tid = threadIdx.x, lane = tid & 31, wave = tid >> 5;
    const int hi16 = (lane < 16) ? 0 : 1;
    const int ntiles = (M + 15) >> 4;
    for (int t0 = blockIdx.x * 8; t0 < ntiles; t0 += gridDim.x * 8) {
        int tile = t0 + wave;
        bool active = tile < ntiles;
        int row = (active ? tile : 0) * 16 + (lane & 15);
        if (row >= M) row = M - 1;
        const __hip_bfloat16* arow = A + (long long)row * KC;
        v8f zero = {0, 0, 0, 0, 0, 0, 0, 0};
        v8f acc[16];
        #pragma unroll
        for (int i = 0; i < 16; ++i) acc[i] = zero;

        auto loadA = [&](FragBF& f, int kg) {
            // 16-bit A layout: lanes<16: K {0..7,16..23}; lanes>=16: {8..15,24..31}
            f.q[0] = *(const uint4*)(arow + kg * 32 + hi16 * 8);
            f.q[1] = *(const uint4*)(arow + kg * 32 + 16 + hi16 * 8);
        };
        auto loadB = [&](FragBF& f, int nb, int kb) {
            // B layout: lanes<16: K {0..7,16..23}; lanes>=16: {8..15,24..31} (N = lane&15)
            const uint4* bp = (const uint4*)(sW + (nb * 16 + (lane & 15)) * 128 +
                                             kb * 32 + hi16 * 8);
            f.q[0] = bp[0]; f.q[1] = bp[2];
        };

        for (int c = 0; c < 4; ++c) {
            __syncthreads();
            stage_chunk(WT, sW, tid, c);
            __syncthreads();
            if (active) {
                FragBF fa, fan;
                loadA(fa, c * 4);
                #pragma unroll
                for (int kb = 0; kb < 4; ++kb) {
                    if (kb < 3) loadA(fan, c * 4 + kb + 1);   // prefetch next A
                    FragBF fb0, fb1;
                    loadB(fb0, 0, kb);
                    #pragma unroll
                    for (int nb = 0; nb < 16; ++nb) {
                        FragBF& cur = (nb & 1) ? fb1 : fb0;
                        FragBF& nxt = (nb & 1) ? fb0 : fb1;
                        if (nb < 15) loadB(nxt, nb + 1, kb);  // double-buffer B
                        acc[nb] = __builtin_amdgcn_wmma_f32_16x16x32_bf16(
                            false, fa.v, false, cur.v, (short)0, acc[nb], false, false);
                    }
                    fa = fan;
                }
            }
        }
        if (active) {
            #pragma unroll
            for (int nb = 0; nb < 16; ++nb) {
                int col = nb * 16 + (lane & 15);
                float b = bias[col];
                #pragma unroll
                for (int v = 0; v < 8; ++v) {
                    int r = tile * 16 + v + hi16 * 8;   // C layout: M=v (+8 for hi half)
                    if (r < M) {
                        float val = fmaxf(acc[nb][v] + b, 0.0f);
                        if (outF) outF[(long long)r * DIM + col] = val;
                        else      outB[(long long)r * DIM + col] = __float2bfloat16(val);
                    }
                }
            }
        }
    }
}

// ------------- edge MLP: out = relu([h[src],h[dst]] @ Wm1 + bm1) @ Wm2 + bm2 -------------
__global__ __launch_bounds__(256) void edge_mlp(
    const __hip_bfloat16* __restrict__ H,      // [NN,256] bf16
    const int* __restrict__ src, const int* __restrict__ dst,
    const __hip_bfloat16* __restrict__ WT,     // Wm1T [256][512] bf16
    const float* __restrict__ bm1, const float* __restrict__ Wm2,
    const float* __restrict__ bm2, float* __restrict__ out, int E)
{
    __shared__ __hip_bfloat16 sW[256 * 128];
    const int tid = threadIdx.x, lane = tid & 31, wave = tid >> 5;
    const int hi16 = (lane < 16) ? 0 : 1;
    const int ntiles = (E + 15) >> 4;
    for (int t0 = blockIdx.x * 8; t0 < ntiles; t0 += gridDim.x * 8) {
        int tile = t0 + wave;
        bool active = tile < ntiles;
        long long e = (long long)(active ? tile : 0) * 16 + (lane & 15);
        if (e >= E) e = E - 1;
        const __hip_bfloat16* rs = H + (long long)src[e] * DIM;  // K 0..255
        const __hip_bfloat16* rd = H + (long long)dst[e] * DIM;  // K 256..511
        v8f zero = {0, 0, 0, 0, 0, 0, 0, 0};
        v8f acc[16];
        #pragma unroll
        for (int i = 0; i < 16; ++i) acc[i] = zero;

        auto loadA = [&](FragBF& f, int kg) {
            const __hip_bfloat16* base = (kg < 8) ? (rs + kg * 32) : (rd + (kg - 8) * 32);
            f.q[0] = *(const uint4*)(base + hi16 * 8);
            f.q[1] = *(const uint4*)(base + 16 + hi16 * 8);
        };
        auto loadB = [&](FragBF& f, int nb, int kb) {
            const uint4* bp = (const uint4*)(sW + (nb * 16 + (lane & 15)) * 128 +
                                             kb * 32 + hi16 * 8);
            f.q[0] = bp[0]; f.q[1] = bp[2];
        };

        for (int c = 0; c < 4; ++c) {
            __syncthreads();
            stage_chunk(WT, sW, tid, c);
            __syncthreads();
            if (active) {
                FragBF fa, fan;
                loadA(fa, c * 4);
                #pragma unroll
                for (int kb = 0; kb < 4; ++kb) {
                    if (kb < 3) loadA(fan, c * 4 + kb + 1);
                    FragBF fb0, fb1;
                    loadB(fb0, 0, kb);
                    #pragma unroll
                    for (int nb = 0; nb < 16; ++nb) {
                        FragBF& cur = (nb & 1) ? fb1 : fb0;
                        FragBF& nxt = (nb & 1) ? fb0 : fb1;
                        if (nb < 15) loadB(nxt, nb + 1, kb);
                        acc[nb] = __builtin_amdgcn_wmma_f32_16x16x32_bf16(
                            false, fa.v, false, cur.v, (short)0, acc[nb], false, false);
                    }
                    fa = fan;
                }
            }
        }
        if (active) {
            // tiny 256->2 projection straight from the WMMA C layout
            float  bv[16]; float2 wv[16];
            #pragma unroll
            for (int nb = 0; nb < 16; ++nb) {
                int n = nb * 16 + (lane & 15);
                bv[nb] = bm1[n];
                wv[nb] = ((const float2*)Wm2)[n];
            }
            #pragma unroll
            for (int v = 0; v < 8; ++v) {
                float p0 = 0.f, p1 = 0.f;
                #pragma unroll
                for (int nb = 0; nb < 16; ++nb) {
                    float h = fmaxf(acc[nb][v] + bv[nb], 0.0f);
                    p0 += h * wv[nb].x;
                    p1 += h * wv[nb].y;
                }
                #pragma unroll
                for (int off = 8; off; off >>= 1) {   // reduce within 16-lane halves
                    p0 += __shfl_xor(p0, off, 32);
                    p1 += __shfl_xor(p1, off, 32);
                }
                long long eo = (long long)tile * 16 + v + hi16 * 8;
                if ((lane & 15) == 0 && eo < E) {
                    out[eo * 2 + 0] = p0 + bm2[0];
                    out[eo * 2 + 1] = p1 + bm2[1];
                }
            }
        }
    }
}

extern "C" void kernel_launch(void* const* d_in, const int* in_sizes, int n_in,
                              void* d_out, int out_size, void* d_ws, size_t ws_size,
                              hipStream_t stream) {
    const float* x   = (const float*)d_in[0];
    const int*   ei  = (const int*)d_in[1];
    const float* W1l = (const float*)d_in[2];
    const float* b1l = (const float*)d_in[3];
    const float* W1r = (const float*)d_in[4];
    const float* W2l = (const float*)d_in[5];
    const float* b2l = (const float*)d_in[6];
    const float* W2r = (const float*)d_in[7];
    const float* Wm1 = (const float*)d_in[8];
    const float* bm1 = (const float*)d_in[9];
    const float* Wm2 = (const float*)d_in[10];
    const float* bm2 = (const float*)d_in[11];
    float* out = (float*)d_out;

    const int* src = ei;
    const int* dst = ei + NE;

    char* ws = (char*)d_ws;
    size_t o = 0;
    auto alloc = [&](size_t bytes) -> char* {
        char* p = ws + o; o += (bytes + 255) & ~(size_t)255; return p;
    };
    float* deg = (float*)alloc((size_t)NN * 4);
    float* agg = (float*)alloc((size_t)NN * DIM * 4);
    float* h1  = (float*)alloc((size_t)NN * DIM * 4);
    __hip_bfloat16* abf  = (__hip_bfloat16*)alloc((size_t)NN * KC * 2);
    __hip_bfloat16* h2   = (__hip_bfloat16*)alloc((size_t)NN * DIM * 2);
    __hip_bfloat16* w1t  = (__hip_bfloat16*)alloc((size_t)KC * DIM * 2);
    __hip_bfloat16* w2t  = (__hip_bfloat16*)alloc((size_t)KC * DIM * 2);
    __hip_bfloat16* wm1t = (__hip_bfloat16*)alloc((size_t)KC * DIM * 2);

    (void)hipMemsetAsync(deg, 0, (size_t)NN * 4, stream);
    (void)hipMemsetAsync(agg, 0, (size_t)NN * DIM * 4, stream);

    deg_kernel<<<(NE + 255) / 256, 256, 0, stream>>>(dst, deg, NE);
    {
        long long tot = (long long)NE * 64;
        scatter_kernel<<<(int)((tot + 255) / 256), 256, 0, stream>>>(x, src, dst, agg, NE);
    }
    pack_wt<<<(DIM * KC + 255) / 256, 256, 0, stream>>>(W1l, W1r, w1t, DIM, KC, DIM);
    pack_wt<<<(DIM * KC + 255) / 256, 256, 0, stream>>>(W2l, W2r, w2t, DIM, KC, DIM);
    pack_wt<<<(DIM * KC + 255) / 256, 256, 0, stream>>>(Wm1, Wm1, wm1t, KC, KC, DIM);
    {
        long long tot = (long long)NN * 64;
        build_a<<<(int)((tot + 255) / 256), 256, 0, stream>>>(agg, deg, x, abf, NN);
    }
    gemm_node<<<391, 256, 0, stream>>>(abf, w1t, b1l, h1, nullptr, NN);

    (void)hipMemsetAsync(agg, 0, (size_t)NN * DIM * 4, stream);
    {
        long long tot = (long long)NE * 64;
        scatter_kernel<<<(int)((tot + 255) / 256), 256, 0, stream>>>(h1, src, dst, agg, NE);
    }
    {
        long long tot = (long long)NN * 64;
        build_a<<<(int)((tot + 255) / 256), 256, 0, stream>>>(agg, deg, h1, abf, NN);
    }
    gemm_node<<<391, 256, 0, stream>>>(abf, w2t, b2l, nullptr, h2, NN);

    edge_mlp<<<1024, 256, 0, stream>>>(h2, src, dst, wm1t, bm1, Wm2, bm2, out, NE);
    (void)in_sizes; (void)n_in; (void)out_size; (void)ws_size;
}
